// MultiHeadAttention_9405978378897
// MI455X (gfx1250) — matrix operand order repaired and checked by reading, not compiled, not run on hardware
//
#include <hip/hip_runtime.h>
#include <hip/hip_bf16.h>
#include <cstdint>

// MI455X / gfx1250, wave32. Matrix math via v_wmma_f32_16x16x32_f16.
// Data movement: global_load_async_to_lds_b128 (ASYNCcnt) + TDM tensor_load_to_lds
// (TENSORcnt, with HW LDS padding) where the toolchain exposes them; guarded
// fallbacks keep the compile clean otherwise.
// Pipeline: cvt -> 3x WMMA GEMM (QKV, double-buffered LDS) -> RoPE ->
//           flash-attention (WMMA + TDM K-tiles) -> WMMA GEMM (Wo).

typedef _Float16 half_t;
typedef __attribute__((ext_vector_type(16))) _Float16 v16h;
typedef __attribute__((ext_vector_type(8)))  float    v8f;
typedef __attribute__((ext_vector_type(4)))  float    f32x4;
typedef __attribute__((ext_vector_type(4)))  unsigned int u32x4;
typedef __attribute__((ext_vector_type(8)))  int      i32x8;
typedef __attribute__((ext_vector_type(4)))  int      i32x4;

#define AS_GLOBAL __attribute__((address_space(1)))
#define AS_LDS    __attribute__((address_space(3)))

#ifndef __has_builtin
#define __has_builtin(x) 0
#endif

#if __has_builtin(__builtin_amdgcn_global_load_async_to_lds_b128)
#define HAS_ASYNC 1
#else
#define HAS_ASYNC 0
#endif

#if __has_builtin(__builtin_amdgcn_tensor_load_to_lds)
#define HAS_TDM 1
#else
#define HAS_TDM 0
#endif

union Frag {
  v16h  v;
  f32x4 f[2];
};

#define D_MODEL 1024
#define SEQ     2048
#define NHEAD   16
#define DK      64

// ---- async copy helpers -----------------------------------------------------
// Builtin proto (probed): (global v4i* src, lds v4i* dst, imm offset, imm cpol).
// Integer round-trips give exact AS1 (64-bit) / AS3 (low-32-bit LDS offset)
// pointer representations without relying on C++ addrspace cast rules.
__device__ __forceinline__ void async_b128(void* lds, const void* glob) {
#if HAS_ASYNC
  AS_GLOBAL i32x4* g = (AS_GLOBAL i32x4*)(unsigned long long)(uintptr_t)glob;
  AS_LDS    i32x4* l = (AS_LDS    i32x4*)(unsigned int)(uintptr_t)lds;
  __builtin_amdgcn_global_load_async_to_lds_b128(g, l, 0, 0);
#else
  *(f32x4*)lds = *(const f32x4*)glob;
#endif
}

__device__ __forceinline__ void wait_async0() {
#if HAS_ASYNC
#if __has_builtin(__builtin_amdgcn_s_wait_asynccnt)
  __builtin_amdgcn_s_wait_asynccnt(0);
#else
  asm volatile("s_wait_asynccnt 0x0" ::: "memory");
#endif
#endif
}

#if HAS_TDM
// TDM: 2D tile (tile_rows x tile_cols halfs) from a row-major f16 tensor into
// LDS, with hardware padding of 4 DWORDs (8 halfs) after every 32 DWORDs
// (64 halfs) -> LDS row stride 72 halfs, matching the WMMA fragment layout.
__device__ __forceinline__ void tdm_load_tile_f16(
    uint32_t lds_off, const half_t* gtile, uint32_t rows_total,
    uint32_t row_stride_halfs, uint32_t tile_rows, uint32_t tile_cols)
{
  uint64_t ga = (uint64_t)(uintptr_t)gtile;
  u32x4 g0;
  g0.x = 1u;                                     // count=1 (valid user D#)
  g0.y = lds_off;                                // lds_addr (bytes)
  g0.z = (uint32_t)ga;                           // global_addr[31:0]
  g0.w = (uint32_t)(ga >> 32) | (2u << 30);      // global_addr[56:32] | type=2
  i32x8 g1;
  // data_size=1 (2B) | pad_enable | pad_interval=4 (32 DW) | pad_amount=3 (4 DW)
  g1[0] = (int)((1u << 16) | (1u << 20) | (4u << 22) | (3u << 25));
  g1[1] = (int)((row_stride_halfs & 0xFFFFu) << 16);                  // tensor_dim0 lo
  g1[2] = (int)(((row_stride_halfs >> 16) & 0xFFFFu) |
                ((rows_total & 0xFFFFu) << 16));                      // dim0 hi | dim1 lo
  g1[3] = (int)(((rows_total >> 16) & 0xFFFFu) |
                ((tile_cols & 0xFFFFu) << 16));                       // dim1 hi | tile_dim0
  g1[4] = (int)(tile_rows & 0xFFFFu);                                 // tile_dim1
  g1[5] = (int)row_stride_halfs;                                      // dim0_stride lo
  g1[6] = 0;
  g1[7] = 0;
  i32x4 gz = {0, 0, 0, 0};
#if defined(__clang_major__) && __clang_major__ >= 23
  i32x8 gz8 = {0, 0, 0, 0, 0, 0, 0, 0};
  __builtin_amdgcn_tensor_load_to_lds(g0, g1, gz, gz, gz8, 0);
#else
  __builtin_amdgcn_tensor_load_to_lds(g0, g1, gz, gz, 0);
#endif
}
#endif

// ---------------- f32 -> f16 (8 elements / thread) ----------------
__global__ __launch_bounds__(256)
void cvt_f16_kernel(const float* __restrict__ in, half_t* __restrict__ out, int n8) {
  int i = blockIdx.x * 256 + threadIdx.x;
  if (i < n8) {
    f32x4 a = *(const f32x4*)(in + (size_t)i * 8);
    f32x4 b = *(const f32x4*)(in + (size_t)i * 8 + 4);
    alignas(16) half_t h[8];
    #pragma unroll
    for (int j = 0; j < 4; ++j) { h[j] = (half_t)a[j]; h[4 + j] = (half_t)b[j]; }
    *(f32x4*)(out + (size_t)i * 8) = *(const f32x4*)h;
  }
}

// ---------------- WMMA GEMM: C[M,N] = A[M,K] @ B[K,N] ----------------
// 256 threads = 8 waves, block tile 128x128, wave tile 32x64 (2x4 WMMA accs).
// Double-buffered LDS: A-slab k+32 via async-to-LDS + B-slab via regs while
// the current slab feeds 8 WMMAs/wave.
template<bool OUT_F32>
__global__ __launch_bounds__(256)
void wmma_gemm_kernel(const half_t* __restrict__ A, const half_t* __restrict__ B,
                      half_t* __restrict__ Ch, float* __restrict__ Cf,
                      int M, int N, int K)
{
  __shared__ alignas(16) half_t sA [2][128 * 40];  // [row][k], stride 40 halfs
  __shared__ alignas(16) half_t sBt[2][128 * 40];  // [n][k] (B^T), stride 40

  const int tid  = threadIdx.x;
  const int wave = tid >> 5, lane = tid & 31;
  const int lm = lane & 15, hi = lane >> 4;
  const int wm = (wave >> 1) * 32;
  const int wn = (wave & 1) * 64;
  const int row0 = blockIdx.y * 128;
  const int col0 = blockIdx.x * 128;
  const int lrow = tid >> 1, lseg = tid & 1;
  (void)M;

  v8f acc[2][4] = {};

  const half_t* gA = A + (size_t)(row0 + lrow) * K + lseg * 16;
  const half_t* gB = B + (size_t)(lseg * 16) * N + (col0 + lrow);
  alignas(16) half_t btmp[16];

  // stage slab k0=0 into buffer 0
  async_b128(&sA[0][lrow * 40 + lseg * 16 + 0], gA + 0);
  async_b128(&sA[0][lrow * 40 + lseg * 16 + 8], gA + 8);
  #pragma unroll
  for (int j = 0; j < 16; ++j) btmp[j] = gB[(size_t)j * N];
  *(f32x4*)&sBt[0][lrow * 40 + lseg * 16 + 0] = *(const f32x4*)&btmp[0];
  *(f32x4*)&sBt[0][lrow * 40 + lseg * 16 + 8] = *(const f32x4*)&btmp[8];
  wait_async0();
  __syncthreads();

  int buf = 0;
  for (int k0 = 0; k0 < K; k0 += 32, buf ^= 1) {
    const int k1 = k0 + 32;
    if (k1 < K) {  // prefetch next slab into the other buffer
      async_b128(&sA[buf ^ 1][lrow * 40 + lseg * 16 + 0], gA + k1 + 0);
      async_b128(&sA[buf ^ 1][lrow * 40 + lseg * 16 + 8], gA + k1 + 8);
      #pragma unroll
      for (int j = 0; j < 16; ++j) btmp[j] = gB[(size_t)(k1 + j) * N];
    }

    // A fragment: row m=lm; K chunks [hi*8,+8) and [hi*8+16,+8)
    Frag af[2], bf[4];
    #pragma unroll
    for (int mt = 0; mt < 2; ++mt) {
      const half_t* p = &sA[buf][(wm + mt * 16 + lm) * 40 + hi * 8];
      af[mt].f[0] = *(const f32x4*)p;
      af[mt].f[1] = *(const f32x4*)(p + 16);
    }
    // B fragment: col n=lm; K chunks [hi*8,+8) and [hi*8+16,+8)
    #pragma unroll
    for (int nt = 0; nt < 4; ++nt) {
      const half_t* p = &sBt[buf][(wn + nt * 16 + lm) * 40 + hi * 8];
      bf[nt].f[0] = *(const f32x4*)p;
      bf[nt].f[1] = *(const f32x4*)(p + 16);
    }
    #pragma unroll
    for (int mt = 0; mt < 2; ++mt)
      #pragma unroll
      for (int nt = 0; nt < 4; ++nt)
        acc[mt][nt] = __builtin_amdgcn_wmma_f32_16x16x32_f16(
            false, af[mt].v, false, bf[nt].v, (short)0, acc[mt][nt], false, false);

    if (k1 < K) {
      *(f32x4*)&sBt[buf ^ 1][lrow * 40 + lseg * 16 + 0] = *(const f32x4*)&btmp[0];
      *(f32x4*)&sBt[buf ^ 1][lrow * 40 + lseg * 16 + 8] = *(const f32x4*)&btmp[8];
      wait_async0();
    }
    __syncthreads();
  }

  // C layout: element (m = r + 8*hi, n = lm) in VGPR r
  #pragma unroll
  for (int mt = 0; mt < 2; ++mt)
    #pragma unroll
    for (int nt = 0; nt < 4; ++nt)
      #pragma unroll
      for (int r = 0; r < 8; ++r) {
        int row = row0 + wm + mt * 16 + hi * 8 + r;
        int col = col0 + wn + nt * 16 + lm;
        float v = acc[mt][nt][r];
        if (OUT_F32) Cf[(size_t)row * N + col] = v;
        else         Ch[(size_t)row * N + col] = (half_t)v;
      }
}

// ---------------- RoPE (in place on f16 Q and K) ----------------
__global__ __launch_bounds__(256)
void rope_kernel(half_t* __restrict__ Q, half_t* __restrict__ Kb) {
  int idx  = blockIdx.x * 256 + threadIdx.x;   // 4096 rows * 16 heads * 32 pairs
  int row  = idx >> 9;
  int rem  = idx & 511;
  int head = rem >> 5;
  int i    = rem & 31;
  int s    = row & (SEQ - 1);
  float inv = __expf((float)i * (-2.0f / (float)DK) * 9.210340371976184f); // ln(1e4)
  float ang = (float)s * inv;
  float c = __cosf(ang), sn = __sinf(ang);
  size_t i1 = (size_t)row * D_MODEL + head * DK + i;
  size_t i2 = i1 + DK / 2;
  float q1 = (float)Q[i1],  q2 = (float)Q[i2];
  Q[i1]  = (half_t)(q1 * c - q2 * sn);
  Q[i2]  = (half_t)(q1 * sn + q2 * c);
  float k1 = (float)Kb[i1], k2 = (float)Kb[i2];
  Kb[i1] = (half_t)(k1 * c - k2 * sn);
  Kb[i2] = (half_t)(k1 * sn + k2 * c);
}

// ---------------- Flash attention: per (batch, head, 64-query tile) ----------------
// 128 threads = 4 waves; each wave owns 16 query rows. Online softmax, f32 O accum.
// K tile fetched by the Tensor Data Mover (wave 0 issues, TENSORcnt wait), with
// hardware LDS padding giving the 72-half row stride the fragments expect.
__global__ __launch_bounds__(128)
void flash_attn_kernel(const half_t* __restrict__ Q, const half_t* __restrict__ Kb,
                       const half_t* __restrict__ Vb, half_t* __restrict__ Ob)
{
  __shared__ alignas(16) half_t sK[64 * 72];       // [key][d]  (= B^T for QK^T)
  __shared__ alignas(16) half_t sV[64 * 72];       // [d][key]  (transposed for PV)
  __shared__ alignas(16) half_t sP[4 * 16 * 72];   // per-wave 16x64 prob tile

  const int tid  = threadIdx.x;
  const int wave = tid >> 5, lane = tid & 31;
  const int lm = lane & 15, hi = lane >> 4;
  const int head  = blockIdx.y, batch = blockIdx.z;
  const int qbase = batch * SEQ + blockIdx.x * 64;
  const int cb    = head * DK;

  Frag qf[2];
  {
    const half_t* qp = Q + (size_t)(qbase + wave * 16 + lm) * D_MODEL + cb + hi * 8;
    qf[0].f[0] = *(const f32x4*)(qp + 0);
    qf[0].f[1] = *(const f32x4*)(qp + 16);
    qf[1].f[0] = *(const f32x4*)(qp + 32);
    qf[1].f[1] = *(const f32x4*)(qp + 48);
  }

  float rm[8], rl[8];
  #pragma unroll
  for (int r = 0; r < 8; ++r) { rm[r] = -3.0e38f; rl[r] = 0.f; }
  v8f oacc[4] = {};

  const int lkey = tid >> 1, lseg = tid & 1;
#if HAS_TDM
  const uint32_t sK_off = (uint32_t)(uintptr_t)&sK[0];
#endif

  for (int kt = 0; kt < SEQ / 64; ++kt) {
    const int keyrow = batch * SEQ + kt * 64 + lkey;
    // ---- K tile ----
#if HAS_TDM
    if (wave == 0)
      tdm_load_tile_f16(sK_off, Kb + (size_t)(batch * SEQ + kt * 64) * D_MODEL + cb,
                        2 * SEQ, D_MODEL, 64, 64);
#else
    {
      const half_t* g = Kb + (size_t)keyrow * D_MODEL + cb + lseg * 32;
      #pragma unroll
      for (int j = 0; j < 4; ++j)
        async_b128(&sK[lkey * 72 + lseg * 32 + j * 8], g + j * 8);
    }
#endif
    // ---- V tile, transposed [d][key] (manual: TDM cannot transpose) ----
    {
      const half_t* g = Vb + (size_t)keyrow * D_MODEL + cb + lseg * 32;
      alignas(16) half_t tv[32];
      #pragma unroll
      for (int j = 0; j < 4; ++j)
        *(f32x4*)&tv[j * 8] = *(const f32x4*)(g + j * 8);
      #pragma unroll
      for (int j = 0; j < 32; ++j)
        sV[(lseg * 32 + j) * 72 + lkey] = tv[j];
    }
#if HAS_TDM
    if (wave == 0) __builtin_amdgcn_s_wait_tensorcnt(0);
#else
    wait_async0();
#endif
    __syncthreads();

    // ---- S = Q @ K^T : 16x64 per wave (4 n-tiles x 2 k-steps) ----
    v8f sc[4] = {};
    #pragma unroll
    for (int nt = 0; nt < 4; ++nt) {
      #pragma unroll
      for (int ks = 0; ks < 2; ++ks) {
        Frag bf;
        const half_t* p = &sK[(nt * 16 + lm) * 72 + ks * 32 + hi * 8];
        bf.f[0] = *(const f32x4*)p;
        bf.f[1] = *(const f32x4*)(p + 16);
        sc[nt] = __builtin_amdgcn_wmma_f32_16x16x32_f16(
            false, qf[ks].v, false, bf.v, (short)0, sc[nt], false, false);
      }
    }

    // ---- online softmax (row m = r + 8*hi across the 16-lane half-wave) ----
    #pragma unroll
    for (int r = 0; r < 8; ++r) {
      float mx = -3.0e38f;
      #pragma unroll
      for (int nt = 0; nt < 4; ++nt) {
        sc[nt][r] *= 0.125f;                       // 1/sqrt(64)
        mx = fmaxf(mx, sc[nt][r]);
      }
      #pragma unroll
      for (int m = 1; m < 16; m <<= 1)
        mx = fmaxf(mx, __shfl_xor(mx, m, 32));
      float nm    = fmaxf(rm[r], mx);
      float alpha = __expf(rm[r] - nm);
      float sum = 0.f;
      #pragma unroll
      for (int nt = 0; nt < 4; ++nt) {
        float pv = __expf(sc[nt][r] - nm);
        sc[nt][r] = pv;
        sum += pv;
      }
      #pragma unroll
      for (int m = 1; m < 16; m <<= 1)
        sum += __shfl_xor(sum, m, 32);
      rl[r] = rl[r] * alpha + sum;
      rm[r] = nm;
      #pragma unroll
      for (int nt = 0; nt < 4; ++nt)
        oacc[nt][r] *= alpha;
    }

    // ---- P: C-layout -> LDS row-major, re-read as A fragments ----
    {
      half_t* pw = &sP[wave * 16 * 72];
      #pragma unroll
      for (int nt = 0; nt < 4; ++nt)
        #pragma unroll
        for (int r = 0; r < 8; ++r)
          pw[(hi * 8 + r) * 72 + nt * 16 + lm] = (half_t)sc[nt][r];
    }
    __syncthreads();

    // ---- O += P @ V ----
    {
      const half_t* pr = &sP[wave * 16 * 72] + lm * 72;
      Frag pf[2];
      #pragma unroll
      for (int ks = 0; ks < 2; ++ks) {
        pf[ks].f[0] = *(const f32x4*)(pr + ks * 32 + hi * 8);
        pf[ks].f[1] = *(const f32x4*)(pr + ks * 32 + hi * 8 + 16);
      }
      #pragma unroll
      for (int nt = 0; nt < 4; ++nt) {
        #pragma unroll
        for (int ks = 0; ks < 2; ++ks) {
          Frag bf;
          const half_t* p = &sV[(nt * 16 + lm) * 72 + ks * 32 + hi * 8];
          bf.f[0] = *(const f32x4*)p;
          bf.f[1] = *(const f32x4*)(p + 16);
          oacc[nt] = __builtin_amdgcn_wmma_f32_16x16x32_f16(
              false, pf[ks].v, false, bf.v, (short)0, oacc[nt], false, false);
        }
      }
    }
    __syncthreads();
  }

  #pragma unroll
  for (int nt = 0; nt < 4; ++nt)
    #pragma unroll
    for (int r = 0; r < 8; ++r) {
      int row = qbase + wave * 16 + hi * 8 + r;
      int col = cb + nt * 16 + lm;
      Ob[(size_t)row * D_MODEL + col] = (half_t)(oacc[nt][r] / rl[r]);
    }
}

// ---------------- host orchestration ----------------
extern "C" void kernel_launch(void* const* d_in, const int* in_sizes, int n_in,
                              void* d_out, int out_size, void* d_ws, size_t ws_size,
                              hipStream_t stream) {
  (void)in_sizes; (void)n_in; (void)out_size; (void)ws_size;
  const float* x  = (const float*)d_in[0];
  const float* Wq = (const float*)d_in[1];
  const float* Wk = (const float*)d_in[3];
  const float* Wv = (const float*)d_in[5];
  const float* Wo = (const float*)d_in[7];
  float* out = (float*)d_out;

  const size_t MB = 1024u * 1024u;
  char* ws = (char*)d_ws;
  half_t* xh  = (half_t*)(ws);             // 8 MB x(f16); reused as attn output
  half_t* wqh = (half_t*)(ws +  8 * MB);   // 2 MB each
  half_t* wkh = (half_t*)(ws + 10 * MB);
  half_t* wvh = (half_t*)(ws + 12 * MB);
  half_t* woh = (half_t*)(ws + 14 * MB);
  half_t* Qh  = (half_t*)(ws + 16 * MB);   // 8 MB each
  half_t* Kh  = (half_t*)(ws + 24 * MB);
  half_t* Vh  = (half_t*)(ws + 32 * MB);
  half_t* Ah  = xh;                        // attn output reuses x-f16 buffer

  const int NX = 2 * SEQ * D_MODEL;        // 4,194,304
  const int NW = D_MODEL * D_MODEL;        // 1,048,576
  cvt_f16_kernel<<<(NX / 8 + 255) / 256, 256, 0, stream>>>(x,  xh,  NX / 8);
  cvt_f16_kernel<<<(NW / 8 + 255) / 256, 256, 0, stream>>>(Wq, wqh, NW / 8);
  cvt_f16_kernel<<<(NW / 8 + 255) / 256, 256, 0, stream>>>(Wk, wkh, NW / 8);
  cvt_f16_kernel<<<(NW / 8 + 255) / 256, 256, 0, stream>>>(Wv, wvh, NW / 8);
  cvt_f16_kernel<<<(NW / 8 + 255) / 256, 256, 0, stream>>>(Wo, woh, NW / 8);

  dim3 gg(D_MODEL / 128, (2 * SEQ) / 128);        // 8 x 32 blocks
  wmma_gemm_kernel<false><<<gg, 256, 0, stream>>>(xh, wqh, Qh, nullptr, 2 * SEQ, D_MODEL, D_MODEL);
  wmma_gemm_kernel<false><<<gg, 256, 0, stream>>>(xh, wkh, Kh, nullptr, 2 * SEQ, D_MODEL, D_MODEL);
  wmma_gemm_kernel<false><<<gg, 256, 0, stream>>>(xh, wvh, Vh, nullptr, 2 * SEQ, D_MODEL, D_MODEL);

  rope_kernel<<<(2 * SEQ * NHEAD * (DK / 2)) / 256, 256, 0, stream>>>(Qh, Kh);

  dim3 ag(SEQ / 64, NHEAD, 2);
  flash_attn_kernel<<<ag, 128, 0, stream>>>(Qh, Kh, Vh, Ah);

  wmma_gemm_kernel<true><<<gg, 256, 0, stream>>>(Ah, woh, nullptr, out, 2 * SEQ, D_MODEL, D_MODEL);
}
